// MlpAttention_74500502716485
// MI455X (gfx1250) — matrix operand order repaired and checked by reading, not compiled, not run on hardware
//
#include <hip/hip_runtime.h>
#include <math.h>

#define B_ 32
#define S_ 4096
#define Q_ 512
#define H_ 512
#define V_ 512

typedef __attribute__((ext_vector_type(16))) _Float16 v16h;
typedef __attribute__((ext_vector_type(8)))  float    v8f;

// ---------------------------------------------------------------------------
// Fast tanh: 1 - 2/(exp(2x)+1).  One v_exp_f32 + one v_rcp_f32.
// Correct limits: x->+inf -> 1, x->-inf -> -1, x=0 -> 0.
// ---------------------------------------------------------------------------
__device__ __forceinline__ float fast_tanh(float x) {
    float e = __expf(2.0f * x);
#if __has_builtin(__builtin_amdgcn_rcpf)
    return 1.0f - 2.0f * __builtin_amdgcn_rcpf(e + 1.0f);
#else
    return 1.0f - 2.0f / (e + 1.0f);
#endif
}

// ---------------------------------------------------------------------------
// Kernel 0: zero the context output region (atomics accumulate into it).
// ---------------------------------------------------------------------------
__global__ __launch_bounds__(256) void zero_ctx_kernel(float* __restrict__ p, int n) {
    int i = blockIdx.x * 256 + threadIdx.x;
    if (i < n) p[i] = 0.0f;
}

// ---------------------------------------------------------------------------
// Kernel 1: pq[B,H] = query[B,Q] @ W_query[Q,H] via v_wmma_f32_16x16x32_f16.
// One wave per 16x16 output tile; M-tiles = 2, N-tiles = 32 -> 64 waves.
// K loop: 16 steps of 32.
// ---------------------------------------------------------------------------
__global__ __launch_bounds__(32) void pq_wmma_kernel(const float* __restrict__ query,
                                                     const float* __restrict__ Wq,
                                                     float* __restrict__ pq) {
    const int tile = blockIdx.x;      // 0..63
    const int mt   = tile >> 5;       // 0..1   (32 rows / 16)
    const int nt   = tile & 31;       // 0..31  (512 cols / 16)
    const int lane = threadIdx.x & 31;
    const int l16  = lane & 15;
    const int hi   = lane >> 4;       // 0 or 1
    const int m    = mt * 16 + l16;   // A row for this lane
    const int n    = nt * 16 + l16;   // B/C/D column for this lane

    v8f c = {};
    for (int k0 = 0; k0 < Q_; k0 += 32) {
        v16h a, b;
#pragma unroll
        for (int j = 0; j < 8; ++j) {
            // A (16x32 f16): VGPR j holds K = 2*(j&3) + 16*(j>>2) + 8*hi, +1
            int ka = k0 + 2 * (j & 3) + 16 * (j >> 2) + 8 * hi;
            a[2 * j]     = (_Float16)query[m * Q_ + ka];
            a[2 * j + 1] = (_Float16)query[m * Q_ + ka + 1];
            // B (32x16 f16): VGPR j holds K = 2*(j&3) + 16*(j>>2) + 8*hi, +1 at column n
            int kb = k0 + 2 * (j & 3) + 16 * (j >> 2) + 8 * hi;
            b[2 * j]     = (_Float16)Wq[kb * H_ + n];
            b[2 * j + 1] = (_Float16)Wq[(kb + 1) * H_ + n];
        }
        c = __builtin_amdgcn_wmma_f32_16x16x32_f16(
                /*neg_a=*/false, a, /*neg_b=*/false, b,
                /*c_mod=*/(short)0, c, /*reuse_a=*/false, /*reuse_b=*/false);
    }
    // C/D f32 layout: VGPR r -> M = mt*16 + r + 8*hi, N = n
#pragma unroll
    for (int r = 0; r < 8; ++r)
        pq[(mt * 16 + r + 8 * hi) * H_ + n] = c[r];
}

// ---------------------------------------------------------------------------
// Kernel 2: raw scores[b,s] = sum_h tanh(pk[b,s,h] + pq[b,h]) * we[h]
// Streams projected_keys (268 MB) once. One wave32 per (b,s) row; each lane
// does 4x float4 loads (512B contiguous per wave access) then a shuffle
// reduction. 8 rows per 256-thread block -> 16384 blocks.
// ---------------------------------------------------------------------------
__global__ __launch_bounds__(256) void scores_kernel(const float* __restrict__ pk,
                                                     const float* __restrict__ pq,
                                                     const float* __restrict__ we,
                                                     const unsigned char* __restrict__ mask,
                                                     float* __restrict__ scores) {
    const int  wave = threadIdx.x >> 5;
    const int  lane = threadIdx.x & 31;
    const long row  = (long)blockIdx.x * 8 + wave;       // 0 .. B*S-1
    const int  b    = (int)(row >> 12);                  // S = 4096

    const float* pkrow = pk + row * H_;
    const float* pqb   = pq + (long)b * H_;

    float acc = 0.0f;
#pragma unroll
    for (int i = 0; i < 4; ++i) {
        int h = i * 128 + lane * 4;
        float4 k4 = *(const float4*)(pkrow + h);
        float4 q4 = *(const float4*)(pqb + h);
        float4 w4 = *(const float4*)(we + h);
        acc += fast_tanh(k4.x + q4.x) * w4.x;
        acc += fast_tanh(k4.y + q4.y) * w4.y;
        acc += fast_tanh(k4.z + q4.z) * w4.z;
        acc += fast_tanh(k4.w + q4.w) * w4.w;
    }
#pragma unroll
    for (int off = 16; off > 0; off >>= 1)
        acc += __shfl_xor(acc, off, 32);

    if (lane == 0)
        scores[row] = mask[row] ? acc : -INFINITY;
}

// ---------------------------------------------------------------------------
// Kernel 3: softmax over S=4096, in place, one 1024-thread block per batch.
// ---------------------------------------------------------------------------
__global__ __launch_bounds__(1024) void softmax_kernel(float* __restrict__ scores) {
    __shared__ float red[32];
    __shared__ float bcast;
    float* sc = scores + (size_t)blockIdx.x * S_;
    const int tid = threadIdx.x, lane = tid & 31, wid = tid >> 5;

    float v0 = sc[tid], v1 = sc[tid + 1024], v2 = sc[tid + 2048], v3 = sc[tid + 3072];

    float m = fmaxf(fmaxf(v0, v1), fmaxf(v2, v3));
#pragma unroll
    for (int off = 16; off > 0; off >>= 1) m = fmaxf(m, __shfl_xor(m, off, 32));
    if (lane == 0) red[wid] = m;
    __syncthreads();
    if (wid == 0) {
        float t = red[lane];
#pragma unroll
        for (int off = 16; off > 0; off >>= 1) t = fmaxf(t, __shfl_xor(t, off, 32));
        if (lane == 0) bcast = t;
    }
    __syncthreads();
    m = bcast;

    v0 = __expf(v0 - m); v1 = __expf(v1 - m); v2 = __expf(v2 - m); v3 = __expf(v3 - m);
    float s = v0 + v1 + v2 + v3;
#pragma unroll
    for (int off = 16; off > 0; off >>= 1) s += __shfl_xor(s, off, 32);
    __syncthreads();                         // red[] reuse fence
    if (lane == 0) red[wid] = s;
    __syncthreads();
    if (wid == 0) {
        float t = red[lane];
#pragma unroll
        for (int off = 16; off > 0; off >>= 1) t += __shfl_xor(t, off, 32);
        if (lane == 0) bcast = t;
    }
    __syncthreads();

    float inv = 1.0f / bcast;
    sc[tid] = v0 * inv; sc[tid + 1024] = v1 * inv;
    sc[tid + 2048] = v2 * inv; sc[tid + 3072] = v3 * inv;
}

// ---------------------------------------------------------------------------
// Kernel 4: context[b,v] += sum_{s in chunk} probs[b,s] * values[b,s,v]
// Streams values (268 MB) once. Grid (B, V/256, S/256) = 1024 blocks so the
// stream has enough concurrency for full HBM bandwidth; probs chunk staged in
// LDS; partial sums combined with global f32 atomics.
// ---------------------------------------------------------------------------
__global__ __launch_bounds__(256) void context_kernel(const float* __restrict__ values,
                                                      const float* __restrict__ probs,
                                                      float* __restrict__ ctx) {
    __shared__ float p[256];
    const int b  = blockIdx.x;
    const int v  = blockIdx.y * 256 + threadIdx.x;
    const int s0 = blockIdx.z * 256;

    p[threadIdx.x] = probs[(long)b * S_ + s0 + threadIdx.x];
    __syncthreads();

    const float* vp = values + ((long)b * S_ + s0) * V_ + v;
    float acc = 0.0f;
#pragma unroll 4
    for (int i = 0; i < 256; ++i)
        acc += p[i] * vp[(long)i * V_];

    atomicAdd(&ctx[b * V_ + v], acc);
}

// ---------------------------------------------------------------------------
// Launch: d_out = [ context (B*V) | scores (B*S) ], d_ws holds pq (B*H f32).
// ---------------------------------------------------------------------------
extern "C" void kernel_launch(void* const* d_in, const int* in_sizes, int n_in,
                              void* d_out, int out_size, void* d_ws, size_t ws_size,
                              hipStream_t stream) {
    const float*         query  = (const float*)d_in[0];
    const float*         pk     = (const float*)d_in[1];
    const float*         values = (const float*)d_in[2];
    const unsigned char* mask   = (const unsigned char*)d_in[3];
    const float*         Wq     = (const float*)d_in[4];
    const float*         we     = (const float*)d_in[5];

    float* ctx    = (float*)d_out;          // [B,V]
    float* scores = ctx + B_ * V_;          // [B,S]
    float* pq     = (float*)d_ws;           // [B,H] scratch (64 KB)

    zero_ctx_kernel<<<(B_ * V_ + 255) / 256, 256, 0, stream>>>(ctx, B_ * V_);
    pq_wmma_kernel<<<64, 32, 0, stream>>>(query, Wq, pq);
    scores_kernel<<<(B_ * S_) / 8, 256, 0, stream>>>(pk, pq, we, mask, scores);
    softmax_kernel<<<B_, 1024, 0, stream>>>(scores);
    context_kernel<<<dim3(B_, V_ / 256, S_ / 256), 256, 0, stream>>>(values, scores, ctx);
}
